// GNet_64811056496761
// MI455X (gfx1250) — matrix operand order repaired and checked by reading, not compiled, not run on hardware
//
#include <hip/hip_runtime.h>
#include <cstdint>

#define BN_EPS 1e-5f

typedef __attribute__((ext_vector_type(16))) __bf16 v16bf;
typedef __attribute__((ext_vector_type(8)))  float  v8f;

union V16BF {
  v16bf v;
  unsigned short u[16];
  uint4 q[2];
};

__device__ __forceinline__ unsigned short f2bf(float x) {
  union { float f; unsigned int u; } c;
  c.f = x;
  unsigned int u = c.u;
  u += 0x7FFFu + ((u >> 16) & 1u);   // round-to-nearest-even
  return (unsigned short)(u >> 16);
}

// ---------------------------------------------------------------- utilities
__global__ void zero_f32(float* __restrict__ p, size_t n) {
  size_t i = (size_t)blockIdx.x * blockDim.x + threadIdx.x;
  size_t stride = (size_t)gridDim.x * blockDim.x;
  for (; i < n; i += stride) p[i] = 0.0f;
}

__global__ void pack_bf16(const float* __restrict__ in, unsigned short* __restrict__ out,
                          size_t n) {
  size_t i = (size_t)blockIdx.x * blockDim.x + threadIdx.x;
  size_t stride = (size_t)gridDim.x * blockDim.x;
  for (; i < n; i += stride) out[i] = f2bf(in[i]);
}

// Pre-pack W [FIN x FOUT] fp32 into WMMA B-operand register order, bf16:
//   Wp[((tileN * (FIN/32) + kblk) * 32 + lane) * 16 + j]
//   = bf16( W[(kblk*32 + (lane>>4)*16 + j) * FOUT + tileN*16 + (lane&15)] )
// so each lane reads its 16 B values as 32 contiguous bytes.
__global__ void pack_weight(const float* __restrict__ W, unsigned short* __restrict__ Wp,
                            int FIN, int FOUT) {
  int total = (FOUT / 16) * (FIN / 32) * 32 * 16;
  int i = blockIdx.x * blockDim.x + threadIdx.x;
  int stride = gridDim.x * blockDim.x;
  int kblks = FIN / 32;
  for (; i < total; i += stride) {
    int j = i & 15;
    int lane = (i >> 4) & 31;
    int blk = i >> 9;                 // tileN * kblks + kblk
    int kblk = blk % kblks;
    int tileN = blk / kblks;
    int k = kblk * 32 + (lane >> 4) * 16 + j;
    int n = tileN * 16 + (lane & 15);
    Wp[i] = f2bf(W[(size_t)k * FOUT + n]);
  }
}

__global__ void deg_count(const int* __restrict__ dst, float* __restrict__ deg, int E) {
  int i = blockIdx.x * blockDim.x + threadIdx.x;
  int stride = gridDim.x * blockDim.x;
  for (; i < E; i += stride) atomicAdd(&deg[dst[i]], 1.0f);
}

__global__ void deg_recip(float* __restrict__ deg, int N) {
  int i = blockIdx.x * blockDim.x + threadIdx.x;
  if (i < N) deg[i] = 1.0f / fmaxf(deg[i], 1.0f);
}

// ------------------------------------------------------- dual GEMM via WMMA
// S = H @ Ws, P = H @ Wn, bf16 inputs (pre-packed), f32 accumulate.
// One wave per 16x16 output tile; A tile shared between the two WMMAs.
template <int FIN, int FOUT>
__global__ __launch_bounds__(256) void gemm_dual_wmma(
    const unsigned short* __restrict__ Hbf,   // [N, FIN] bf16 row-major
    const unsigned short* __restrict__ Wps,   // packed Ws
    const unsigned short* __restrict__ Wpn,   // packed Wn
    float* __restrict__ S, float* __restrict__ P, int N) {
  const int lane  = threadIdx.x & 31;
  const int wave  = threadIdx.x >> 5;
  const int tileM = blockIdx.x * 8 + wave;
  const int m0    = tileM * 16;
  if (m0 >= N) return;                        // wave-uniform (EXEC all-1s for WMMA)
  const int half = lane >> 4;
  const int l15  = lane & 15;

  // A row base: 16-byte aligned (FIN and k offsets are multiples of 8 bf16)
  const uint4* __restrict__ arow =
      (const uint4*)(Hbf + (size_t)(m0 + l15) * FIN);
  // Packed B base for this column tile: lane's 16 values are contiguous
  const int kblks = FIN / 32;
  const uint4* __restrict__ bs_base =
      (const uint4*)(Wps + (((size_t)blockIdx.y * kblks) * 32 + lane) * 16);
  const uint4* __restrict__ bn_base =
      (const uint4*)(Wpn + (((size_t)blockIdx.y * kblks) * 32 + lane) * 16);

  v8f cs = {};
  v8f cp = {};

#pragma unroll
  for (int kb = 0; kb < kblks; ++kb) {
    V16BF a, bs, bn;
    // A 16x32: lane<16 -> K 0-15 ; lane>=16 -> K 16-31 (contiguous, matches packed B)
    const int qa = kb * 4 + half * 2;         // uint4 index of first 8-bf16 run
    a.q[0] = arow[qa];
    a.q[1] = arow[qa + 1];
    // B: 32 bytes contiguous per lane per k-block
    bs.q[0] = bs_base[kb * 64 + 0];           // 64 uint4 per k-block (32 lanes x 2)
    bs.q[1] = bs_base[kb * 64 + 1];
    bn.q[0] = bn_base[kb * 64 + 0];
    bn.q[1] = bn_base[kb * 64 + 1];
    cs = __builtin_amdgcn_wmma_f32_16x16x32_bf16(false, a.v, false, bs.v,
                                                 (short)0, cs, false, false);
    cp = __builtin_amdgcn_wmma_f32_16x16x32_bf16(false, a.v, false, bn.v,
                                                 (short)0, cp, false, false);
  }

  const int n0 = blockIdx.y * 16;
#pragma unroll
  for (int i = 0; i < 8; ++i) {
    const int r = m0 + half * 8 + i;
    S[(size_t)r * FOUT + n0 + l15] = cs[i];
    P[(size_t)r * FOUT + n0 + l15] = cp[i];
  }
}

// -------------------------------------------------- edge scatter (agg += P[src] at dst)
template <int LOGF>
__global__ void edge_scatter(const float* __restrict__ P, const int* __restrict__ src,
                             const int* __restrict__ dst, float* __restrict__ agg,
                             size_t total) {
  const int F = 1 << LOGF;
  size_t i = (size_t)blockIdx.x * blockDim.x + threadIdx.x;
  size_t stride = (size_t)gridDim.x * blockDim.x;
  for (; i < total; i += stride) {
    size_t e = i >> LOGF;
    int f = (int)(i & (F - 1));
    int s = src[e];
    int d = dst[e];
    atomicAdd(&agg[((size_t)d << LOGF) + f], P[((size_t)s << LOGF) + f]);
  }
}

// ------------------- combine (fc_self + mean_neigh + bias) + column sum/sumsq (F=64)
__global__ __launch_bounds__(256) void combine_stats64(
    const float* __restrict__ S, const float* __restrict__ agg,
    const float* __restrict__ rdeg, const float* __restrict__ bias,
    float* __restrict__ Hmid, float* __restrict__ colsum,
    float* __restrict__ colsq, int N) {
  const int F = 64, G = 4;                 // 256 threads = 4 row-groups x 64 cols
  const int tid = threadIdx.x;
  const int f = tid & 63;
  const int g = tid >> 6;
  float lsum = 0.0f, lsq = 0.0f;
  for (int row = blockIdx.x * G + g; row < N; row += gridDim.x * G) {
    size_t idx = (size_t)row * F + f;
    float h = S[idx] + agg[idx] * rdeg[row] + bias[f];
    Hmid[idx] = h;
    lsum += h;
    lsq += h * h;
  }
  __shared__ float ssum[256];
  __shared__ float ssq[256];
  ssum[tid] = lsum;
  ssq[tid] = lsq;
  __syncthreads();
  if (g < 2) { ssum[tid] += ssum[tid + 2 * F]; ssq[tid] += ssq[tid + 2 * F]; }
  __syncthreads();
  if (g == 0) {
    atomicAdd(&colsum[f], ssum[f] + ssum[f + F]);
    atomicAdd(&colsq[f], ssq[f] + ssq[f + F]);
  }
}

// ---------------- BN apply (+optional ReLU), F=64; emits bf16 for the next GEMM
__global__ void bn_apply64(const float* __restrict__ Hmid, const float* __restrict__ colsum,
                           const float* __restrict__ colsq, const float* __restrict__ gamma,
                           const float* __restrict__ beta,
                           unsigned short* __restrict__ Hbf, int N, int relu) {
  size_t total = (size_t)N * 64;
  size_t i = (size_t)blockIdx.x * blockDim.x + threadIdx.x;
  size_t stride = (size_t)gridDim.x * blockDim.x;
  const float invN = 1.0f / (float)N;
  for (; i < total; i += stride) {
    int f = (int)(i & 63);
    float mean = colsum[f] * invN;
    float var = colsq[f] * invN - mean * mean;     // biased variance
    float h = (Hmid[i] - mean) * rsqrtf(var + BN_EPS) * gamma[f] + beta[f];
    if (relu) h = fmaxf(h, 0.0f);
    Hbf[i] = f2bf(h);
  }
}

// ---------------------------------------------- final layer combine (F=16, no BN)
__global__ void combine_out16(const float* __restrict__ S, const float* __restrict__ agg,
                              const float* __restrict__ rdeg, const float* __restrict__ bias,
                              float* __restrict__ out, int N) {
  size_t total = (size_t)N * 16;
  size_t i = (size_t)blockIdx.x * blockDim.x + threadIdx.x;
  size_t stride = (size_t)gridDim.x * blockDim.x;
  for (; i < total; i += stride) {
    int f = (int)(i & 15);
    int row = (int)(i >> 4);
    out[i] = S[i] + agg[i] * rdeg[row] + bias[f];
  }
}

extern "C" void kernel_launch(void* const* d_in, const int* in_sizes, int n_in,
                              void* d_out, int out_size, void* d_ws, size_t ws_size,
                              hipStream_t stream) {
  (void)n_in; (void)out_size; (void)ws_size;
  const float* x   = (const float*)d_in[0];
  const int*   src = (const int*)d_in[1];
  const int*   dst = (const int*)d_in[2];
  const float *Ws1 = (const float*)d_in[3],  *Wn1 = (const float*)d_in[4],
              *b1  = (const float*)d_in[5],  *g1  = (const float*)d_in[6],
              *be1 = (const float*)d_in[7];
  const float *Ws2 = (const float*)d_in[8],  *Wn2 = (const float*)d_in[9],
              *b2  = (const float*)d_in[10], *g2  = (const float*)d_in[11],
              *be2 = (const float*)d_in[12];
  const float *Ws3 = (const float*)d_in[13], *Wn3 = (const float*)d_in[14],
              *b3  = (const float*)d_in[15], *g3  = (const float*)d_in[16],
              *be3 = (const float*)d_in[17];
  const float *Ws4 = (const float*)d_in[18], *Wn4 = (const float*)d_in[19],
              *b4  = (const float*)d_in[20];

  const int N = in_sizes[0] / 128;
  const int E = in_sizes[1];

  // workspace carve-out
  char* ws = (char*)d_ws;
  size_t off = 0;
  auto carve = [&](size_t bytes) -> void* {
    void* p = (void*)(ws + off);
    off = (off + bytes + 255) & ~(size_t)255;
    return p;
  };
  float* S      = (float*)carve((size_t)N * 64 * 4);
  float* P      = (float*)carve((size_t)N * 64 * 4);
  float* agg    = (float*)carve((size_t)N * 64 * 4);
  float* Hmid   = (float*)carve((size_t)N * 64 * 4);
  float* rdeg   = (float*)carve((size_t)N * 4);
  float* cstats = (float*)carve(128 * 4);      // [0:64) colsum, [64:128) colsumsq
  float* colsum = cstats;
  float* colsq  = cstats + 64;
  unsigned short* xbf = (unsigned short*)carve((size_t)N * 128 * 2);
  unsigned short* Hbf = (unsigned short*)carve((size_t)N * 64 * 2);
  unsigned short* Wp1s = (unsigned short*)carve(128 * 64 * 2);
  unsigned short* Wp1n = (unsigned short*)carve(128 * 64 * 2);
  unsigned short* Wp2s = (unsigned short*)carve(64 * 64 * 2);
  unsigned short* Wp2n = (unsigned short*)carve(64 * 64 * 2);
  unsigned short* Wp3s = (unsigned short*)carve(64 * 64 * 2);
  unsigned short* Wp3n = (unsigned short*)carve(64 * 64 * 2);
  unsigned short* Wp4s = (unsigned short*)carve(64 * 16 * 2);
  unsigned short* Wp4n = (unsigned short*)carve(64 * 16 * 2);

  const dim3 blk(256);
  auto gs = [](size_t total) -> unsigned {
    size_t b = (total + 255) / 256;
    if (b > (size_t)(1u << 20)) b = (size_t)(1u << 20);
    return (unsigned)b;
  };

  const int tilesM = (N + 15) / 16;
  const dim3 ggrid64((tilesM + 7) / 8, 4);     // FOUT=64 -> 4 column tiles
  const dim3 ggrid16((tilesM + 7) / 8, 1);     // FOUT=16 -> 1 column tile
  const size_t n64 = (size_t)N * 64;
  const size_t n16 = (size_t)N * 16;
  const size_t e64 = (size_t)E << 6;
  const size_t e16 = (size_t)E << 4;

  // one-time prep: degrees, bf16 input, packed bf16 weights
  zero_f32<<<gs(N), blk, 0, stream>>>(rdeg, (size_t)N);
  deg_count<<<gs(E), blk, 0, stream>>>(dst, rdeg, E);
  deg_recip<<<(unsigned)((N + 255) / 256), blk, 0, stream>>>(rdeg, N);
  pack_bf16<<<gs((size_t)N * 128), blk, 0, stream>>>(x, xbf, (size_t)N * 128);
  pack_weight<<<32, blk, 0, stream>>>(Ws1, Wp1s, 128, 64);
  pack_weight<<<32, blk, 0, stream>>>(Wn1, Wp1n, 128, 64);
  pack_weight<<<16, blk, 0, stream>>>(Ws2, Wp2s, 64, 64);
  pack_weight<<<16, blk, 0, stream>>>(Wn2, Wp2n, 64, 64);
  pack_weight<<<16, blk, 0, stream>>>(Ws3, Wp3s, 64, 64);
  pack_weight<<<16, blk, 0, stream>>>(Wn3, Wp3n, 64, 64);
  pack_weight<<<4, blk, 0, stream>>>(Ws4, Wp4s, 64, 16);
  pack_weight<<<4, blk, 0, stream>>>(Wn4, Wp4n, 64, 16);

  // ---- layer 1: SAGE(128->64) + BN ----
  zero_f32<<<gs(n64), blk, 0, stream>>>(agg, n64);
  zero_f32<<<1, 128, 0, stream>>>(cstats, 128);
  gemm_dual_wmma<128, 64><<<ggrid64, blk, 0, stream>>>(xbf, Wp1s, Wp1n, S, P, N);
  edge_scatter<6><<<gs(e64), blk, 0, stream>>>(P, src, dst, agg, e64);
  combine_stats64<<<1024, blk, 0, stream>>>(S, agg, rdeg, b1, Hmid, colsum, colsq, N);
  bn_apply64<<<gs(n64), blk, 0, stream>>>(Hmid, colsum, colsq, g1, be1, Hbf, N, 0);

  // ---- layer 2: SAGE(64->64) + BN + ReLU ----
  zero_f32<<<gs(n64), blk, 0, stream>>>(agg, n64);
  zero_f32<<<1, 128, 0, stream>>>(cstats, 128);
  gemm_dual_wmma<64, 64><<<ggrid64, blk, 0, stream>>>(Hbf, Wp2s, Wp2n, S, P, N);
  edge_scatter<6><<<gs(e64), blk, 0, stream>>>(P, src, dst, agg, e64);
  combine_stats64<<<1024, blk, 0, stream>>>(S, agg, rdeg, b2, Hmid, colsum, colsq, N);
  bn_apply64<<<gs(n64), blk, 0, stream>>>(Hmid, colsum, colsq, g2, be2, Hbf, N, 1);

  // ---- layer 3: SAGE(64->64) + BN + ReLU ----
  zero_f32<<<gs(n64), blk, 0, stream>>>(agg, n64);
  zero_f32<<<1, 128, 0, stream>>>(cstats, 128);
  gemm_dual_wmma<64, 64><<<ggrid64, blk, 0, stream>>>(Hbf, Wp3s, Wp3n, S, P, N);
  edge_scatter<6><<<gs(e64), blk, 0, stream>>>(P, src, dst, agg, e64);
  combine_stats64<<<1024, blk, 0, stream>>>(S, agg, rdeg, b3, Hmid, colsum, colsq, N);
  bn_apply64<<<gs(n64), blk, 0, stream>>>(Hmid, colsum, colsq, g3, be3, Hbf, N, 1);

  // ---- layer 4: SAGE(64->16), no BN ----
  zero_f32<<<gs(n16), blk, 0, stream>>>(agg, n16);
  gemm_dual_wmma<64, 16><<<ggrid16, blk, 0, stream>>>(Hbf, Wp4s, Wp4n, S, P, N);
  edge_scatter<4><<<gs(e16), blk, 0, stream>>>(P, src, dst, agg, e16);
  combine_out16<<<gs(n16), blk, 0, stream>>>(S, agg, rdeg, b4, (float*)d_out, N);
}
